// Global_Model_Merged_Deep_BSDE_79319456023265
// MI455X (gfx1250) — matrix operand order repaired and checked by reading, not compiled, not run on hardware
//
#include <hip/hip_runtime.h>

// ---------------------------------------------------------------------------
// Deep-BSDE rollout, MI455X (gfx1250, wave32, WMMA f16 path).
// y_final[b] = y0 + sum_t ( DT * sum_d z^2 + sum_d z * dw )   (LAM = 1)
// Time steps are additive -> parallelize over (row-tile, time-chunk).
// ---------------------------------------------------------------------------

#define DIM    100
#define NDIS   50
#define HID    110
#define BATCH  8192
#define DT_VAL 0.02f

#define BM     128               // batch rows per workgroup
#define NTHR   256               // 8 waves of 32
#define LDSTR  136               // padded LDS row stride (halves): 272B = 68 dwords
#define KTILES 4                 // K padded to 128 = 4 * 32
#define NTILES 7                 // N padded to 112 = 7 * 16
#define NCHUNK 10                // time chunks
#define TSTEPS (NDIS / NCHUNK)   // 5 steps per chunk

typedef _Float16 half8  __attribute__((ext_vector_type(8)));
typedef _Float16 half16 __attribute__((ext_vector_type(16)));
typedef float    f32x8  __attribute__((ext_vector_type(8)));

// A-matrix fragment (16x32 f16): lane L (r = L&15 is row, h = L>>4 picks K-phase)
// halves[0..7] = K[h*8 .. h*8+7], halves[8..15] = K[16+h*8 .. 16+h*8+7]
__device__ __forceinline__ half16 load_a_frag(const _Float16* p) {
  half8 lo = *(const half8*)(p);
  half8 hi = *(const half8*)(p + 16);
  return __builtin_shufflevector(lo, hi, 0, 1, 2, 3, 4, 5, 6, 7,
                                         8, 9, 10, 11, 12, 13, 14, 15);
}

// B-matrix fragment (32x16 f16): lane L holds column n = L&15,
// one contiguous 32B chunk of Wt[n][k]; unused: gemm_tiles loads B via load_a_frag (A's K order).
__device__ __forceinline__ half16 load_b_frag(const _Float16* p) {
  return *(const half16*)(p);
}

// One [16 x 128] x [128 x 112] GEMM slice per wave: 4 K-tiles x 7 N-tiles.
// All fragment loads for one K-tile are hoisted ahead of the 7-WMMA burst so
// the ds_load_b128 cluster issues before a single s_wait covers them all
// (instead of a full s_wait_dscnt 0 drain per WMMA).
__device__ __forceinline__ void gemm_tiles(const _Float16* __restrict__ A,
                                           const _Float16* __restrict__ Wt,
                                           f32x8 acc[NTILES], int w, int lane) {
  const int r = lane & 15;
  const int h = lane >> 4;
  const f32x8 zero = {0.f, 0.f, 0.f, 0.f, 0.f, 0.f, 0.f, 0.f};
#pragma unroll
  for (int nt = 0; nt < NTILES; ++nt) acc[nt] = zero;
#pragma unroll
  for (int kt = 0; kt < KTILES; ++kt) {
    half16 a = load_a_frag(&A[(w * 16 + r) * LDSTR + kt * 32 + h * 8]);
    half16 b[NTILES];
#pragma unroll
    for (int nt = 0; nt < NTILES; ++nt)
      b[nt] = load_a_frag(&Wt[(nt * 16 + r) * LDSTR + kt * 32 + h * 8]);
#pragma unroll
    for (int nt = 0; nt < NTILES; ++nt)
      acc[nt] = __builtin_amdgcn_wmma_f32_16x16x32_f16(
          false, a, false, b[nt], (short)0, acc[nt], false, false);
  }
}

// C/D layout: lane L holds column n = L&15, VGPR r -> row (L>>4)*8 + r.
__device__ __forceinline__ void store_relu(const f32x8 acc[NTILES],
                                           _Float16* __restrict__ dst,
                                           int w, int lane) {
  const int n  = lane & 15;
  const int hb = (lane >> 4) * 8;
#pragma unroll
  for (int nt = 0; nt < NTILES; ++nt) {
#pragma unroll
    for (int r = 0; r < 8; ++r) {
      float v = acc[nt][r];
      v = v > 0.f ? v : 0.f;
      dst[(w * 16 + hb + r) * LDSTR + nt * 16 + n] = (_Float16)v;
    }
  }
}

// Wt[n*LDSTR + k] = (k < Kv && n < Nv) ? W[k*Nv + n] : 0   (B operand, transposed)
__device__ __forceinline__ void fill_wt(_Float16* __restrict__ dst,
                                        const float* __restrict__ src,
                                        int Kv, int Nv, int tid) {
  for (int i = tid; i < 112 * 128; i += NTHR) {
    int n = i % 112;           // consecutive tids -> consecutive n -> coalesced src
    int k = i / 112;
    float v = (k < Kv && n < Nv) ? src[k * Nv + n] : 0.f;
    dst[n * LDSTR + k] = (_Float16)v;
  }
}

__global__ __launch_bounds__(NTHR)
void bsde_rollout_kernel(const float* __restrict__ dw, const float* __restrict__ x,
                         const float* __restrict__ W1, const float* __restrict__ W2,
                         const float* __restrict__ W3, float* __restrict__ ws) {
  __shared__ _Float16 sWt1[112 * LDSTR];
  __shared__ _Float16 sWt2[112 * LDSTR];
  __shared__ _Float16 sWt3[112 * LDSTR];
  __shared__ _Float16 sA[BM * LDSTR];   // inp (step start) / h2 (after layer 2)
  __shared__ _Float16 sH[BM * LDSTR];   // h1
  __shared__ float    sY[BM];

  const int tid     = (int)threadIdx.x;
  const int lane    = tid & 31;
  const int w       = tid >> 5;
  const int rowbase = (int)blockIdx.x * BM;
  const int chunk   = (int)blockIdx.y;

  // One-time zero of the K-padding columns (112..127) and everything else.
  for (int i = tid; i < BM * LDSTR; i += NTHR) {
    sA[i] = (_Float16)0.f;
    sH[i] = (_Float16)0.f;
  }
  if (tid < BM) sY[tid] = 0.f;

  fill_wt(sWt1, W1, DIM + 1, HID, tid);
  fill_wt(sWt2, W2, HID,     HID, tid);
  fill_wt(sWt3, W3, HID,     DIM, tid);

  f32x8 acc[NTILES];

  for (int i = 0; i < TSTEPS; ++i) {
    const int   t    = chunk * TSTEPS + i;
    const float tval = (float)t * DT_VAL;

    __syncthreads();  // covers init on i==0; protects sA from prior GEMM3 reads

    // Build A = [t, x[:, :, t]] as f16, cols 0..111 (101..111 are zeros).
    for (int idx = tid; idx < BM * 112; idx += NTHR) {
      const int m = idx / 112;
      const int c = idx % 112;
      float v = 0.f;
      size_t base = (size_t)(rowbase + m) * DIM;
      if (c == 0) {
        v = tval;
      } else if (c <= DIM) {
        v = x[(base + (c - 1)) * NDIS + t];
        if (i + 1 < TSTEPS)
          __builtin_prefetch(&x[(base + (c - 1)) * NDIS + t + 1], 0, 0);
      }
      sA[m * LDSTR + c] = (_Float16)v;
    }
    __syncthreads();

    gemm_tiles(sA, sWt1, acc, w, lane);   // h1 = relu(inp @ W1)
    store_relu(acc, sH, w, lane);
    __syncthreads();

    gemm_tiles(sH, sWt2, acc, w, lane);   // h2 = relu(h1 @ W2)
    store_relu(acc, sA, w, lane);
    __syncthreads();

    gemm_tiles(sA, sWt3, acc, w, lane);   // zraw = h2 @ W3  (z = zraw / DIM)

    // Epilogue: s = sum_col DT*z^2 + z*dw ; reduce over the 16 lanes per row.
    {
      const int n  = lane & 15;
      const int hb = (lane >> 4) * 8;
#pragma unroll
      for (int r = 0; r < 8; ++r) {
        const int    m    = w * 16 + hb + r;
        const size_t brow = (size_t)(rowbase + m) * DIM;
        float s = 0.f;
#pragma unroll
        for (int nt = 0; nt < NTILES; ++nt) {
          const int col = nt * 16 + n;
          if (col < DIM) {
            const float z   = acc[nt][r] * (1.0f / (float)DIM);
            const float dwv = dw[(brow + col) * NDIS + t];
            s += DT_VAL * z * z + z * dwv;   // -DT*f + z.dw, LAM = 1
          }
        }
        s += __shfl_xor(s, 1);
        s += __shfl_xor(s, 2);
        s += __shfl_xor(s, 4);
        s += __shfl_xor(s, 8);
        if (n == 0) sY[m] += s;   // unique (wave, half-wave) owner per row
      }
    }
  }

  __syncthreads();
  for (int m = tid; m < BM; m += NTHR)
    ws[(size_t)chunk * BATCH + (size_t)(rowbase + m)] = sY[m];
}

__global__ __launch_bounds__(256)
void bsde_finalize_kernel(const float* __restrict__ ws, const float* __restrict__ y0,
                          float* __restrict__ out) {
  const int b = (int)blockIdx.x * 256 + (int)threadIdx.x;
  if (b < BATCH) {
    float s = y0[0];
#pragma unroll
    for (int c = 0; c < NCHUNK; ++c) s += ws[(size_t)c * BATCH + b];
    out[b] = s;
  }
}

extern "C" void kernel_launch(void* const* d_in, const int* in_sizes, int n_in,
                              void* d_out, int out_size, void* d_ws, size_t ws_size,
                              hipStream_t stream) {
  (void)in_sizes; (void)n_in; (void)out_size; (void)ws_size;
  const float* dw = (const float*)d_in[0];
  const float* x  = (const float*)d_in[1];
  const float* W1 = (const float*)d_in[2];
  const float* W2 = (const float*)d_in[3];
  const float* W3 = (const float*)d_in[4];
  const float* y0 = (const float*)d_in[5];
  float* ws  = (float*)d_ws;            // NCHUNK * BATCH floats = 320 KB
  float* out = (float*)d_out;

  dim3 grid(BATCH / BM, NCHUNK);        // 64 row-tiles x 10 time-chunks
  bsde_rollout_kernel<<<grid, NTHR, 0, stream>>>(dw, x, W1, W2, W3, ws);
  bsde_finalize_kernel<<<(BATCH + 255) / 256, 256, 0, stream>>>(ws, y0, out);
}
